// VanilaLSTM_65017214926939
// MI455X (gfx1250) — compile-verified, matrix operand order checked by reading, not run on hardware
//
#include <hip/hip_runtime.h>

#define BATCH  64
#define SEQ    512
#define INPUT  512
#define HIDDEN 1024
#define OUTPUT 512
#define FOURH  (4 * HIDDEN)

typedef __attribute__((ext_vector_type(16))) __bf16 v16bf;
typedef __attribute__((ext_vector_type(8)))  __bf16 v8bf;
typedef __attribute__((ext_vector_type(8)))  float  v8f;

// Scheduler fence: no instructions may be reordered across this point.
// Used to pin the load-prefetch / WMMA-consume pipeline stages.
__device__ __forceinline__ void sched_fence() { __builtin_amdgcn_sched_barrier(0); }

// Load one WMMA bf16 fragment (A or B, same per-lane layout):
// lane holds elements [off, off+8) and [off+16, off+24) of its row,
// where off = k0 + (lane>>4)*8. Two 16-byte dwordx4 loads.
__device__ __forceinline__ v16bf frag_load(const __bf16* __restrict__ rowbase, int off) {
    union { v16bf v; v8bf h[2]; } u;
    u.h[0] = *reinterpret_cast<const v8bf*>(rowbase + off);
    u.h[1] = *reinterpret_cast<const v8bf*>(rowbase + off + 16);
    return u.v;
}

__device__ __forceinline__ v8f wmma_bf16(v16bf a, v16bf b, v8f c) {
    // (neg_a, A, neg_b, B, c_mod, C, reuse_a, reuse_b)
    return __builtin_amdgcn_wmma_f32_16x16x32_bf16(false, a, false, b, (short)0, c, false, false);
}

__device__ __forceinline__ float fsig(float x) {
    return 1.0f / (1.0f + __expf(-x));
}
__device__ __forceinline__ float ftanh(float x) {
    float xx = fminf(fmaxf(x, -15.0f), 15.0f);
    float e  = __expf(2.0f * xx);
    return (e - 1.0f) / (e + 1.0f);
}

// Software-pipelined 4-gate GEMM over one K-phase.
// Depth-2 prefetch with a 3-buffer rotation: loads for stage s+2 are in
// flight while stage s is consumed, so each 4-WMMA group waits only for
// loadcnt <= 20 (two stages outstanding). sched_barrier(0) fences keep the
// scheduler from sinking the prefetch loads below the WMMAs.
template<int K>
__device__ __forceinline__ void gemm4_pipelined(
        const __bf16* __restrict__ a,
        const __bf16* __restrict__ bF, const __bf16* __restrict__ bI,
        const __bf16* __restrict__ bC, const __bf16* __restrict__ bO,
        int ko, v8f& accF, v8f& accI, v8f& accC, v8f& accO) {
    constexpr int S = K / 32;                    // number of k-stages
    v16bf A[3], Fb[3], Ib[3], Cb[3], Ob[3];
    #pragma unroll
    for (int s = 0; s < 2 && s < S; ++s) {       // prologue: stages 0,1
        A[s]  = frag_load(a,  s * 32 + ko);
        Fb[s] = frag_load(bF, s * 32 + ko);
        Ib[s] = frag_load(bI, s * 32 + ko);
        Cb[s] = frag_load(bC, s * 32 + ko);
        Ob[s] = frag_load(bO, s * 32 + ko);
    }
    sched_fence();
    #pragma unroll
    for (int s = 0; s < S; ++s) {
        if (s + 2 < S) {                         // prefetch stage s+2
            const int nb = (s + 2) % 3;
            A[nb]  = frag_load(a,  (s + 2) * 32 + ko);
            Fb[nb] = frag_load(bF, (s + 2) * 32 + ko);
            Ib[nb] = frag_load(bI, (s + 2) * 32 + ko);
            Cb[nb] = frag_load(bC, (s + 2) * 32 + ko);
            Ob[nb] = frag_load(bO, (s + 2) * 32 + ko);
        }
        sched_fence();
        const int cu = s % 3;                    // consume stage s
        accF = wmma_bf16(A[cu], Fb[cu], accF);
        accI = wmma_bf16(A[cu], Ib[cu], accI);
        accC = wmma_bf16(A[cu], Cb[cu], accC);
        accO = wmma_bf16(A[cu], Ob[cu], accO);
        sched_fence();
    }
}

// Depth-2 pipelined single-accumulator GEMM (for the output projection).
template<int K>
__device__ __forceinline__ void gemm1_pipelined(
        const __bf16* __restrict__ a, const __bf16* __restrict__ b,
        int ko, v8f& acc) {
    constexpr int S = K / 32;
    v16bf A[3], B[3];
    #pragma unroll
    for (int s = 0; s < 2 && s < S; ++s) {
        A[s] = frag_load(a, s * 32 + ko);
        B[s] = frag_load(b, s * 32 + ko);
    }
    sched_fence();
    #pragma unroll
    for (int s = 0; s < S; ++s) {
        if (s + 2 < S) {
            const int nb = (s + 2) % 3;
            A[nb] = frag_load(a, (s + 2) * 32 + ko);
            B[nb] = frag_load(b, (s + 2) * 32 + ko);
        }
        sched_fence();
        const int cu = s % 3;
        acc = wmma_bf16(A[cu], B[cu], acc);
        sched_fence();
    }
}

// ---------------- preprocessing kernels ----------------

__global__ __launch_bounds__(256) void k_cvt_bf16(const float* __restrict__ s,
                                                  __bf16* __restrict__ d, int n) {
    int i = blockIdx.x * blockDim.x + threadIdx.x;
    if (i < n) d[i] = (__bf16)s[i];
}

__global__ __launch_bounds__(256) void k_cpy_f32(const float* __restrict__ s,
                                                 float* __restrict__ d, int n) {
    int i = blockIdx.x * blockDim.x + threadIdx.x;
    if (i < n) d[i] = s[i];
}

// x: (B, T, I) f32  ->  xbf: (T, B, I) bf16   (per-timestep A panels contiguous)
__global__ __launch_bounds__(256) void k_cvt_x(const float* __restrict__ x,
                                               __bf16* __restrict__ xbf) {
    int idx = blockIdx.x * blockDim.x + threadIdx.x;   // over T*B*I
    int i   = idx % INPUT;
    int rem = idx / INPUT;
    int b   = rem % BATCH;
    int t   = rem / BATCH;
    xbf[idx] = (__bf16)x[((size_t)b * SEQ + t) * INPUT + i];
}

__global__ __launch_bounds__(256) void k_zero_state(__bf16* __restrict__ h0,
                                                    __bf16* __restrict__ h1,
                                                    float* __restrict__ c, int n) {
    int i = blockIdx.x * blockDim.x + threadIdx.x;
    if (i < n) { h0[i] = (__bf16)0.0f; h1[i] = (__bf16)0.0f; c[i] = 0.0f; }
}

// ---------------- recurrent step ----------------
// One wave computes a 16(batch) x 16(hidden) tile of ALL FOUR gates,
// fused K-loop over [x_t (K=512) | h_{t-1} (K=1024)], then applies the
// LSTM cell update. hIn/hOut are ping-pong buffers (cross-wave WAR safety).
__global__ __launch_bounds__(256)
void lstm_step(const __bf16* __restrict__ xbf,   // [SEQ][BATCH][INPUT]
               const __bf16* __restrict__ Wx,    // [4H][INPUT]  rows: f,i,c,o blocks
               const __bf16* __restrict__ Wh,    // [4H][HIDDEN]
               const float*  __restrict__ bh,    // [4H]
               const __bf16* __restrict__ hIn,   // [BATCH][HIDDEN] (t-1)
               __bf16*       __restrict__ hOut,  // [BATCH][HIDDEN] (t)
               float*        __restrict__ cst,   // [BATCH][HIDDEN]
               int t) {
    const int lane = threadIdx.x & 31;
    const int wave = threadIdx.x >> 5;
    const int w    = blockIdx.x * 8 + wave;      // 0..255
    const int tb   = w >> 6;                     // batch tile 0..3
    const int tn   = w & 63;                     // hidden tile 0..63
    const int mrow = lane & 15;
    const int ko   = (lane >> 4) << 3;           // 0 or 8

    const __bf16* aX = xbf + ((size_t)t * BATCH + (tb * 16 + mrow)) * INPUT;
    const __bf16* aH = hIn + (size_t)(tb * 16 + mrow) * HIDDEN;

    const int jn = tn * 16 + mrow;               // weight row within gate block
    const __bf16* bFx = Wx + (size_t)(0 * HIDDEN + jn) * INPUT;
    const __bf16* bIx = Wx + (size_t)(1 * HIDDEN + jn) * INPUT;
    const __bf16* bCx = Wx + (size_t)(2 * HIDDEN + jn) * INPUT;
    const __bf16* bOx = Wx + (size_t)(3 * HIDDEN + jn) * INPUT;
    const __bf16* bFh = Wh + (size_t)(0 * HIDDEN + jn) * HIDDEN;
    const __bf16* bIh = Wh + (size_t)(1 * HIDDEN + jn) * HIDDEN;
    const __bf16* bCh = Wh + (size_t)(2 * HIDDEN + jn) * HIDDEN;
    const __bf16* bOh = Wh + (size_t)(3 * HIDDEN + jn) * HIDDEN;

    // Hoisted epilogue operands: issued first, complete (in-order) long
    // before the epilogue needs them; zero cost against the pipeline waits.
    const int ngl  = tn * 16 + (lane & 15);
    const int mofs = (lane >> 4) * 8;
    const float bF = bh[0 * HIDDEN + ngl];
    const float bI = bh[1 * HIDDEN + ngl];
    const float bC = bh[2 * HIDDEN + ngl];
    const float bO = bh[3 * HIDDEN + ngl];
    float cOld[8];
    #pragma unroll
    for (int r = 0; r < 8; ++r)
        cOld[r] = cst[(size_t)(tb * 16 + r + mofs) * HIDDEN + ngl];

    v8f aF = {}, aI = {}, aC = {}, aO = {};

    gemm4_pipelined<INPUT >(aX, bFx, bIx, bCx, bOx, ko, aF, aI, aC, aO);
    gemm4_pipelined<HIDDEN>(aH, bFh, bIh, bCh, bOh, ko, aF, aI, aC, aO);

    // epilogue: C/D layout -> element (m = r + 8*(lane>>4), n = lane&15)
    #pragma unroll
    for (int r = 0; r < 8; ++r) {
        size_t idx = (size_t)(tb * 16 + r + mofs) * HIDDEN + ngl;
        float gf = fsig(aF[r] + bF);
        float gi = fsig(aI[r] + bI);
        float gc = ftanh(aC[r] + bC);
        float go = fsig(aO[r] + bO);
        float cn = gf * cOld[r] + gi * gc;
        cst[idx]  = cn;
        hOut[idx] = (__bf16)(go * ftanh(cn));
    }
}

// ---------------- output projection: out = h @ Wout^T + bout ----------------
__global__ __launch_bounds__(256)
void lstm_out(const __bf16* __restrict__ hbf,   // [BATCH][HIDDEN]
              const __bf16* __restrict__ Wo,    // [OUTPUT][HIDDEN]
              const float*  __restrict__ bout,  // [OUTPUT]
              float*        __restrict__ out) { // [BATCH][OUTPUT]
    const int lane = threadIdx.x & 31;
    const int wave = threadIdx.x >> 5;
    const int w    = blockIdx.x * 8 + wave;      // 0..127
    const int tb   = w >> 5;                     // 0..3
    const int to   = w & 31;                     // 0..31
    const int mrow = lane & 15;
    const int ko   = (lane >> 4) << 3;

    const __bf16* aH = hbf + (size_t)(tb * 16 + mrow) * HIDDEN;
    const __bf16* bW = Wo  + (size_t)(to * 16 + mrow) * HIDDEN;

    const int ogl  = to * 16 + (lane & 15);
    const float bb = bout[ogl];

    v8f acc = {};
    gemm1_pipelined<HIDDEN>(aH, bW, ko, acc);

    const int mofs = (lane >> 4) * 8;
    #pragma unroll
    for (int r = 0; r < 8; ++r) {
        int b = tb * 16 + r + mofs;
        out[(size_t)b * OUTPUT + ogl] = acc[r] + bb;
    }
}

// ---------------- host orchestration ----------------

extern "C" void kernel_launch(void* const* d_in, const int* in_sizes, int n_in,
                              void* d_out, int out_size, void* d_ws, size_t ws_size,
                              hipStream_t stream) {
    (void)in_sizes; (void)n_in; (void)out_size; (void)ws_size;
    const float* x    = (const float*)d_in[0];
    const float* WxS[4] = { (const float*)d_in[1], (const float*)d_in[4],
                            (const float*)d_in[7], (const float*)d_in[10] };
    const float* WhS[4] = { (const float*)d_in[2], (const float*)d_in[5],
                            (const float*)d_in[8], (const float*)d_in[11] };
    const float* bhS[4] = { (const float*)d_in[3], (const float*)d_in[6],
                            (const float*)d_in[9], (const float*)d_in[12] };
    const float* Wout = (const float*)d_in[13];
    const float* bout = (const float*)d_in[14];
    float* out = (float*)d_out;

    uint8_t* p = (uint8_t*)d_ws;
    auto take = [&](size_t bytes) { uint8_t* q = p; p += (bytes + 255) & ~(size_t)255; return q; };
    __bf16* Wxb  = (__bf16*)take((size_t)FOURH * INPUT  * 2);  //  4 MB
    __bf16* Whb  = (__bf16*)take((size_t)FOURH * HIDDEN * 2);  //  8 MB
    __bf16* Wob  = (__bf16*)take((size_t)OUTPUT * HIDDEN * 2); //  1 MB
    float*  bhb  = (float*) take((size_t)FOURH * 4);           // 16 KB
    float*  cst  = (float*) take((size_t)BATCH * HIDDEN * 4);  // 256 KB
    __bf16* h0   = (__bf16*)take((size_t)BATCH * HIDDEN * 2);  // 128 KB
    __bf16* h1   = (__bf16*)take((size_t)BATCH * HIDDEN * 2);  // 128 KB
    __bf16* xbf  = (__bf16*)take((size_t)SEQ * BATCH * INPUT * 2); // 32 MB

    // 1) pack weights/biases into bf16 concatenated [f,i,c,o] blocks (L2-resident)
    for (int g = 0; g < 4; ++g) {
        int nx = HIDDEN * INPUT;
        k_cvt_bf16<<<(nx + 255) / 256, 256, 0, stream>>>(WxS[g], Wxb + (size_t)g * nx, nx);
        int nh = HIDDEN * HIDDEN;
        k_cvt_bf16<<<(nh + 255) / 256, 256, 0, stream>>>(WhS[g], Whb + (size_t)g * nh, nh);
        k_cpy_f32<<<(HIDDEN + 255) / 256, 256, 0, stream>>>(bhS[g], bhb + (size_t)g * HIDDEN, HIDDEN);
    }
    {
        int nw = OUTPUT * HIDDEN;
        k_cvt_bf16<<<(nw + 255) / 256, 256, 0, stream>>>(Wout, Wob, nw);
    }

    // 2) x -> bf16, transposed to [T][B][I]
    {
        int nx = SEQ * BATCH * INPUT;
        k_cvt_x<<<nx / 256, 256, 0, stream>>>(x, xbf);
    }

    // 3) zero initial state
    {
        int ns = BATCH * HIDDEN;
        k_zero_state<<<(ns + 255) / 256, 256, 0, stream>>>(h0, h1, cst, ns);
    }

    // 4) 512 recurrent steps, ping-pong h buffers (t even: h0->h1, t odd: h1->h0)
    for (int t = 0; t < SEQ; ++t) {
        const __bf16* hIn = (t & 1) ? h1 : h0;
        __bf16*       hOt = (t & 1) ? h0 : h1;
        lstm_step<<<32, 256, 0, stream>>>(xbf, Wxb, Whb, bhb, hIn, hOt, cst, t);
    }

    // 5) output projection from final h (t=511 wrote h0)
    lstm_out<<<16, 256, 0, stream>>>(h0, Wob, bout, out);
}
